// GGNN_13357348291350
// MI455X (gfx1250) — compile-verified, matrix operand order checked by reading, not run on hardware
//
#include <hip/hip_runtime.h>
#include <hip/hip_bf16.h>
#include <math.h>

#define H   512
#define H2  256
#define H4  128
#define NOE 151
#define NOP 51
#define NE  1024
#define NP  16384
#define EMB 300
#define TSTEPS 3

typedef __attribute__((ext_vector_type(16))) __bf16 v16bf;
typedef __attribute__((ext_vector_type(8)))  __bf16 v8bf;
typedef __attribute__((ext_vector_type(8)))  float  v8f;

// ---------------------------------------------------------------------------
// Operand conversion: f32 -> bf16 (A, row-major [M,K] unchanged)
// ---------------------------------------------------------------------------
__global__ void ggnn_f32_to_bf16(const float* __restrict__ src, __bf16* __restrict__ dst, int n) {
    int i = blockIdx.x * blockDim.x + threadIdx.x;
    if (i < n) dst[i] = (__bf16)src[i];
}
// Weight conversion: produce Bt[Npad, K] bf16 with K contiguous.
// transB=0: src is [K,N] row-major -> Bt[n,k] = src[k*N+n]
// transB=1: src is [N,K] row-major -> Bt[n,k] = src[n*K+k]
// rows n in [N, Npad) are zero.
__global__ void ggnn_wconv(const float* __restrict__ src, __bf16* __restrict__ dst,
                           int K, int N, int Npad, int transB) {
    int i = blockIdx.x * blockDim.x + threadIdx.x;
    if (i >= Npad * K) return;
    int k = i % K, n = i / K;
    float v = 0.f;
    if (n < N) v = transB ? src[(size_t)n * K + k] : src[(size_t)k * N + n];
    dst[i] = (__bf16)v;
}

// ---------------------------------------------------------------------------
// WMMA GEMM: C[M,Nout] = act(A[M,K] @ Bt[Npad,K]^T + bias)
// A, Bt pre-converted bf16. Requires: M % 128 == 0, K % 32 == 0, Npad % 64 == 0.
// Wave tile 32x64: 2 A-frags x 4 B-frags -> 8 accumulators; 4 waves/block = 128 rows.
// Fragment layout per ISA (16-bit A 16x32): lane l (l16=l&15, half=l>>4):
//   elems 0..7  -> K = kb..kb+7        (kb = 8*half)
//   elems 8..15 -> K = 16+kb..16+kb+7
// i.e. two contiguous 8xbf16 (16B) chunks -> two global_load_b128, no guards.
// C/D: vgpr r -> M = r + 8*half, N = l16.
// ---------------------------------------------------------------------------
__global__ void ggnn_wmma_gemm(const __bf16* __restrict__ A, const __bf16* __restrict__ Bt,
                               const float* __restrict__ bias, float* __restrict__ C,
                               int M, int Nout, int K, int relu)
{
    const int lane = threadIdx.x & 31;
    const int wave = threadIdx.x >> 5;
    const int n0 = blockIdx.x * 64;
    const int m0 = blockIdx.y * 128 + wave * 32;
    const int half = lane >> 4;
    const int l16  = lane & 15;
    const int kb   = half * 8;

    v8f acc[2][4] = {};
    const __bf16* ap0 = A + (size_t)(m0 + l16) * K + kb;
    const __bf16* ap1 = A + (size_t)(m0 + 16 + l16) * K + kb;
    const __bf16* bp0 = Bt + (size_t)(n0 +  0 + l16) * K + kb;
    const __bf16* bp1 = Bt + (size_t)(n0 + 16 + l16) * K + kb;
    const __bf16* bp2 = Bt + (size_t)(n0 + 32 + l16) * K + kb;
    const __bf16* bp3 = Bt + (size_t)(n0 + 48 + l16) * K + kb;

    for (int k0 = 0; k0 < K; k0 += 32) {
        v8bf a0l = *(const v8bf*)(ap0 + k0), a0h = *(const v8bf*)(ap0 + k0 + 16);
        v8bf a1l = *(const v8bf*)(ap1 + k0), a1h = *(const v8bf*)(ap1 + k0 + 16);
        v8bf b0l = *(const v8bf*)(bp0 + k0), b0h = *(const v8bf*)(bp0 + k0 + 16);
        v8bf b1l = *(const v8bf*)(bp1 + k0), b1h = *(const v8bf*)(bp1 + k0 + 16);
        v8bf b2l = *(const v8bf*)(bp2 + k0), b2h = *(const v8bf*)(bp2 + k0 + 16);
        v8bf b3l = *(const v8bf*)(bp3 + k0), b3h = *(const v8bf*)(bp3 + k0 + 16);
        v16bf a0 = __builtin_shufflevector(a0l, a0h, 0,1,2,3,4,5,6,7,8,9,10,11,12,13,14,15);
        v16bf a1 = __builtin_shufflevector(a1l, a1h, 0,1,2,3,4,5,6,7,8,9,10,11,12,13,14,15);
        v16bf b0 = __builtin_shufflevector(b0l, b0h, 0,1,2,3,4,5,6,7,8,9,10,11,12,13,14,15);
        v16bf b1 = __builtin_shufflevector(b1l, b1h, 0,1,2,3,4,5,6,7,8,9,10,11,12,13,14,15);
        v16bf b2 = __builtin_shufflevector(b2l, b2h, 0,1,2,3,4,5,6,7,8,9,10,11,12,13,14,15);
        v16bf b3 = __builtin_shufflevector(b3l, b3h, 0,1,2,3,4,5,6,7,8,9,10,11,12,13,14,15);
        acc[0][0] = __builtin_amdgcn_wmma_f32_16x16x32_bf16(false, a0, false, b0, (short)0, acc[0][0], false, false);
        acc[0][1] = __builtin_amdgcn_wmma_f32_16x16x32_bf16(false, a0, false, b1, (short)0, acc[0][1], false, false);
        acc[0][2] = __builtin_amdgcn_wmma_f32_16x16x32_bf16(false, a0, false, b2, (short)0, acc[0][2], false, false);
        acc[0][3] = __builtin_amdgcn_wmma_f32_16x16x32_bf16(false, a0, false, b3, (short)0, acc[0][3], false, false);
        acc[1][0] = __builtin_amdgcn_wmma_f32_16x16x32_bf16(false, a1, false, b0, (short)0, acc[1][0], false, false);
        acc[1][1] = __builtin_amdgcn_wmma_f32_16x16x32_bf16(false, a1, false, b1, (short)0, acc[1][1], false, false);
        acc[1][2] = __builtin_amdgcn_wmma_f32_16x16x32_bf16(false, a1, false, b2, (short)0, acc[1][2], false, false);
        acc[1][3] = __builtin_amdgcn_wmma_f32_16x16x32_bf16(false, a1, false, b3, (short)0, acc[1][3], false, false);
    }
#pragma unroll
    for (int mf = 0; mf < 2; ++mf)
#pragma unroll
        for (int nf = 0; nf < 4; ++nf)
#pragma unroll
            for (int r = 0; r < 8; ++r) {
                int mm = m0 + mf * 16 + half * 8 + r;
                int nn = n0 + nf * 16 + l16;
                if (nn < Nout) {
                    float v = acc[mf][nf][r];
                    if (bias) v += bias[nn];
                    if (relu) v = fmaxf(v, 0.f);
                    C[(size_t)mm * Nout + nn] = v;
                }
            }
}

// Naive GEMM for tiny ontology-side matrices (M <= 151).
__global__ void ggnn_naive_gemm(const float* __restrict__ A, const float* __restrict__ B,
                                const float* __restrict__ bias, float* __restrict__ C,
                                int M, int N, int K, int relu, int transB)
{
    size_t i = (size_t)blockIdx.x * blockDim.x + threadIdx.x;
    if (i >= (size_t)M * N) return;
    int m = (int)(i / N), n = (int)(i % N);
    float s = bias ? bias[n] : 0.f;
    for (int k = 0; k < K; ++k)
        s += A[(size_t)m * K + k] * (transB ? B[(size_t)n * K + k] : B[(size_t)k * N + n]);
    if (relu) s = fmaxf(s, 0.f);
    C[i] = s;
}

// ---------------- elementwise -------------------------------------------------
__global__ void ggnn_sigadd(const float* a, const float* b, float* o, int n) {
    int i = blockIdx.x * blockDim.x + threadIdx.x;
    if (i < n) o[i] = 1.f / (1.f + __expf(-(a[i] + b[i])));
}
__global__ void ggnn_mul(const float* a, const float* b, float* o, int n) {
    int i = blockIdx.x * blockDim.x + threadIdx.x;
    if (i < n) o[i] = a[i] * b[i];
}
__global__ void ggnn_grufin(const float* z, const float* x5, const float* x6, float* h, int n) {
    int i = blockIdx.x * blockDim.x + threadIdx.x;
    if (i < n) {
        float zz = z[i];
        float hh = tanhf(x5[i] + x6[i]);
        h[i] = (1.f - zz) * h[i] + zz * hh;
    }
}

// l2-normalize groups of 128 floats (dim=2, eps=1e-4): one wave per group.
__global__ void ggnn_l2norm128(float* x, int groups) {
    int g = blockIdx.x;
    if (g >= groups) return;
    float* p = x + (size_t)g * 128;
    int l = threadIdx.x; // 0..31
    float v0 = p[l], v1 = p[l + 32], v2 = p[l + 64], v3 = p[l + 96];
    float s = v0 * v0 + v1 * v1 + v2 * v2 + v3 * v3;
    for (int o = 16; o > 0; o >>= 1) s += __shfl_down(s, o, 32);
    s = __shfl(s, 0, 32);
    float inv = 1.f / sqrtf(fmaxf(s, 1e-4f));
    p[l] = v0 * inv; p[l + 32] = v1 * inv; p[l + 64] = v2 * inv; p[l + 96] = v3 * inv;
}

__global__ void ggnn_rowsoftmax(const float* in, float* out, int rows, int cols) {
    int r = blockIdx.x * blockDim.x + threadIdx.x;
    if (r >= rows) return;
    const float* x = in + (size_t)r * cols;
    float* y = out + (size_t)r * cols;
    float mx = -1e30f;
    for (int i = 0; i < cols; ++i) mx = fmaxf(mx, x[i]);
    float s = 0.f;
    for (int i = 0; i < cols; ++i) { float e = __expf(x[i] - mx); y[i] = e; s += e; }
    float inv = 1.f / s;
    for (int i = 0; i < cols; ++i) y[i] *= inv;
}

// ---------------- adjacency (row-norm folded in) -----------------------------
__global__ void ggnn_colsum_den(const float* adj, float* den, int T, int S, int R) {
    int i = blockIdx.x * blockDim.x + threadIdx.x;
    if (i >= T * R) return;
    int t = i / R, r = i % R;
    float s = 0.f;
    for (int ss = 0; ss < S; ++ss) s += adj[((size_t)t * S + ss) * R + r];
    den[i] = s;
}
// inc[r, slot0+t, :] = (sum_s adj[t,s,r]*ms[s,:]) / max(den[t,r],1)
__global__ void ggnn_adj_einsum(const float* adj, const float* den, const float* ms,
                                float* inc, int S, int R, int T, int slot0, int slots) {
    int tid = blockIdx.x * blockDim.x + threadIdx.x;
    if (tid >= R * T * 128) return;
    int f = tid & 127, rt = tid >> 7;
    int t = rt % T, r = rt / T;
    float acc = 0.f;
    for (int s = 0; s < S; ++s)
        acc += adj[((size_t)t * S + s) * R + r] * ms[(size_t)s * 128 + f];
    acc /= fmaxf(den[t * R + r], 1.f);
    inc[((size_t)r * slots + slot0 + t) * 128 + f] = acc;
}

// ---------------- one-hot pred<->entity routing ------------------------------
__global__ void ggnn_counts(const int* rel, float* cnt_s, float* cnt_o, int P) {
    int p = blockIdx.x * blockDim.x + threadIdx.x;
    if (p >= P) return;
    atomicAdd(&cnt_s[rel[2 * p]], 1.f);
    atomicAdd(&cnt_o[rel[2 * p + 1]], 1.f);
}
__global__ void ggnn_scatter_subobj(const int* rel, const float* ms_ip,
                                    float* acc_s, float* acc_o, int P) {
    int tid = blockIdx.x * blockDim.x + threadIdx.x;
    if (tid >= P * 128) return;
    int f = tid & 127, p = tid >> 7;
    float v = ms_ip[tid];
    atomicAdd(&acc_s[(size_t)rel[2 * p] * 128 + f], v);
    atomicAdd(&acc_o[(size_t)rel[2 * p + 1] * 128 + f], v);
}
__global__ void ggnn_scale_inc_ie(const float* acc_s, const float* acc_o,
                                  const float* cnt_s, const float* cnt_o,
                                  float* inc_ie, int E) {
    int tid = blockIdx.x * blockDim.x + threadIdx.x;
    if (tid >= E * 128) return;
    int f = tid & 127, e = tid >> 7;
    inc_ie[((size_t)e * 3 + 0) * 128 + f] = acc_s[tid] / fmaxf(cnt_s[e], 1.f);
    inc_ie[((size_t)e * 3 + 1) * 128 + f] = acc_o[tid] / fmaxf(cnt_o[e], 1.f);
}
__global__ void ggnn_gather_ip01(const int* rel, const float* ms_ie, float* inc_ip, int P) {
    int tid = blockIdx.x * blockDim.x + threadIdx.x;
    if (tid >= P * 128) return;
    int f = tid & 127, p = tid >> 7;
    inc_ip[((size_t)p * 3 + 0) * 128 + f] = ms_ie[(size_t)rel[2 * p] * 128 + f];
    inc_ip[((size_t)p * 3 + 1) * 128 + f] = ms_ie[(size_t)rel[2 * p + 1] * 128 + f];
}
// inc_ip[:,2,:] = img2ont[P,51] @ ms_op[51,128]
__global__ void ggnn_ip_slot2(const float* img2ont, const float* ms_op, float* inc_ip, int P) {
    int tid = blockIdx.x * blockDim.x + threadIdx.x;
    if (tid >= P * 128) return;
    int f = tid & 127, p = tid >> 7;
    float acc = 0.f;
    for (int o = 0; o < NOP; ++o)
        acc += img2ont[(size_t)p * NOP + o] * ms_op[(size_t)o * 128 + f];
    inc_ip[((size_t)p * 3 + 2) * 128 + f] = acc;
}
// inc_op[:,4,:] = img2ont^T[51,P] @ ms_ip[P,128]
__global__ void ggnn_op_slot4(const float* img2ont, const float* ms_ip, float* inc_op, int P) {
    int o = blockIdx.x, f = threadIdx.x;
    float acc = 0.f;
    for (int p = 0; p < P; ++p)
        acc += img2ont[(size_t)p * NOP + o] * ms_ip[(size_t)p * 128 + f];
    inc_op[((size_t)o * 5 + 4) * 128 + f] = acc;
}

// ---------------- top-5 img-ent -> ont-ent bridge (computed once) ------------
__global__ void ggnn_topk(const float* logits, int* idx5, float* val5, float* colsum, int E) {
    int e = blockIdx.x * blockDim.x + threadIdx.x;
    if (e >= E) return;
    const float* L = logits + (size_t)e * NOE;
    float mx = -1e30f;
    for (int i = 1; i < NOE; ++i) mx = fmaxf(mx, L[i]);
    float sum = 0.f;
    for (int i = 1; i < NOE; ++i) sum += __expf(L[i] - mx);
    float bv[5]; int bi[5];
    for (int j = 0; j < 5; ++j) { bv[j] = -1.f; bi[j] = 0; }
    for (int i = 0; i < NOE; ++i) {
        float v = (i == 0) ? 0.f : (__expf(L[i] - mx) / sum);
        for (int j = 0; j < 5; ++j) {
            if (v > bv[j]) {
                for (int k = 4; k > j; --k) { bv[k] = bv[k - 1]; bi[k] = bi[k - 1]; }
                bv[j] = v; bi[j] = i;
                break;
            }
        }
    }
    for (int j = 0; j < 5; ++j) {
        idx5[e * 5 + j] = bi[j];
        val5[e * 5 + j] = bv[j];
        atomicAdd(&colsum[bi[j]], bv[j]);
    }
}
// inc_ie[:,2,:] = i2o_ent_raw[E,151] @ ms_oe[151,128]  (5 nonzeros per row)
__global__ void ggnn_gather_ie2(const int* idx5, const float* val5,
                                const float* ms_oe, float* inc_ie, int E) {
    int tid = blockIdx.x * blockDim.x + threadIdx.x;
    if (tid >= E * 128) return;
    int f = tid & 127, e = tid >> 7;
    float acc = 0.f;
    for (int k = 0; k < 5; ++k)
        acc += val5[e * 5 + k] * ms_oe[(size_t)idx5[e * 5 + k] * 128 + f];
    inc_ie[((size_t)e * 3 + 2) * 128 + f] = acc;
}
// acc_oe[o,:] += val * ms_ie[e,:]  (i2o_ent^T @ ms_ie via scatter)
__global__ void ggnn_scatter_i2o(const int* idx5, const float* val5,
                                 const float* ms_ie, float* acc_oe, int E) {
    int tid = blockIdx.x * blockDim.x + threadIdx.x;
    if (tid >= E * 128) return;
    int f = tid & 127, e = tid >> 7;
    float m = ms_ie[tid];
    for (int k = 0; k < 5; ++k)
        atomicAdd(&acc_oe[(size_t)idx5[e * 5 + k] * 128 + f], val5[e * 5 + k] * m);
}
__global__ void ggnn_scale_acc_oe(const float* acc_oe, const float* colsum, float* inc_oe) {
    int tid = blockIdx.x * blockDim.x + threadIdx.x;
    if (tid >= NOE * 128) return;
    int f = tid & 127, o = tid >> 7;
    inc_oe[((size_t)o * 7 + 6) * 128 + f] = acc_oe[tid] / fmaxf(colsum[o], 1.f);
}

// ---------------------------------------------------------------------------
extern "C" void kernel_launch(void* const* d_in, const int* in_sizes, int n_in,
                              void* d_out, int out_size, void* d_ws, size_t ws_size,
                              hipStream_t stream) {
    (void)in_sizes; (void)n_in; (void)out_size; (void)ws_size;
    const int*   rel      = (const int*)d_in[0];
    const float* logits   = (const float*)d_in[1];
    const float* obj_fm   = (const float*)d_in[2];
    const float* vr       = (const float*)d_in[3];
    const float* emb_ent  = (const float*)d_in[4];
    const float* emb_pred = (const float*)d_in[5];
    const float* adj_e2e  = (const float*)d_in[6];
    const float* adj_e2p  = (const float*)d_in[7];
    const float* adj_p2e  = (const float*)d_in[8];
    const float* adj_p2p  = (const float*)d_in[9];
    const float* Wie = (const float*)d_in[10]; const float* bie = (const float*)d_in[11];
    const float* Wip = (const float*)d_in[12]; const float* bip = (const float*)d_in[13];
    const float* sW1 = (const float*)d_in[14]; const float* sb1 = (const float*)d_in[15];
    const float* sW2 = (const float*)d_in[16]; const float* sb2 = (const float*)d_in[17];
    const float* roeW1 = (const float*)d_in[18]; const float* roeB1 = (const float*)d_in[19];
    const float* roeW2 = (const float*)d_in[20]; const float* roeB2 = (const float*)d_in[21];
    const float* ropW1 = (const float*)d_in[22]; const float* ropB1 = (const float*)d_in[23];
    const float* ropW2 = (const float*)d_in[24]; const float* ropB2 = (const float*)d_in[25];
    const float* rieW1 = (const float*)d_in[26]; const float* rieB1 = (const float*)d_in[27];
    const float* rieW2 = (const float*)d_in[28]; const float* rieB2 = (const float*)d_in[29];
    const float* ripW1 = (const float*)d_in[30]; const float* ripB1 = (const float*)d_in[31];
    const float* ripW2 = (const float*)d_in[32]; const float* ripB2 = (const float*)d_in[33];
    const float* gruW  = (const float*)d_in[34]; const float* gruB  = (const float*)d_in[35];
    const float* oiW1 = (const float*)d_in[36]; const float* oiB1 = (const float*)d_in[37];
    const float* oiW2 = (const float*)d_in[38]; const float* oiB2 = (const float*)d_in[39];
    const float* ooW1 = (const float*)d_in[40]; const float* ooB1 = (const float*)d_in[41];
    const float* ooW2 = (const float*)d_in[42]; const float* ooB2 = (const float*)d_in[43];
    float* out = (float*)d_out; // [NP, NOP]

    // workspace carve-up (32B-aligned blocks so b128 loads stay aligned)
    float* ws = (float*)d_ws;
    size_t off = 0;
    auto alloc = [&](size_t n) { float* p = ws + off; off += (n + 7) & ~(size_t)7; return p; };
    float* nodes_oe = alloc((size_t)NOE * H);
    float* nodes_op = alloc((size_t)NOP * H);
    float* nodes_ie = alloc((size_t)NE * H);
    float* nodes_ip = alloc((size_t)NP * H);
    float* img2ont  = alloc((size_t)NP * NOP);
    float* ms_oe = alloc((size_t)NOE * H4);
    float* ms_op = alloc((size_t)NOP * H4);
    float* ms_ie = alloc((size_t)NE * H4);
    float* ms_ip = alloc((size_t)NP * H4);
    float* inc_oe = alloc((size_t)NOE * 7 * H4);
    float* inc_op = alloc((size_t)NOP * 5 * H4);
    float* inc_ie = alloc((size_t)NE * 3 * H4);
    float* inc_ip = alloc((size_t)NP * 3 * H4);
    float* mr_oe = alloc((size_t)NOE * H);
    float* mr_op = alloc((size_t)NOP * H);
    float* mr_ie = alloc((size_t)NE * H);
    float* XA = alloc((size_t)NE * H);
    float* XB = alloc((size_t)NE * H);
    float* XC = alloc((size_t)NE * H);
    float* XD = alloc((size_t)NE * H);
    float* S0 = alloc((size_t)NP * H);
    float* S1 = alloc((size_t)NP * H);
    float* S2 = alloc((size_t)NP * H);
    float* S3 = alloc((size_t)NP * H);
    __bf16* Abf = (__bf16*)alloc((size_t)NP * H / 2);   // M*K bf16 staging (max 16384x512)
    __bf16* Bbf = (__bf16*)alloc((size_t)H * H / 2);    // Npad*K bf16 staging (max 512x512)
    float* den_e2e = alloc(4 * NOE);
    float* den_e2p = alloc(2 * NOP);
    float* den_p2e = alloc(2 * NOE);
    float* den_p2p = alloc(2 * NOP);
    float* cnt_s = alloc(NE);
    float* cnt_o = alloc(NE);
    float* colsum = alloc(NOE);
    float* val5 = alloc(NE * 5);
    int*   idx5 = (int*)alloc(NE * 5);
    float* acc_s  = alloc((size_t)NE * H4);
    float* acc_o  = alloc((size_t)NE * H4);
    float* acc_oe = alloc((size_t)NOE * H4);

    auto ew = [&](int n) { return dim3((n + 255) / 256); };
    auto gemm = [&](const float* A, const float* B, const float* bias, float* C,
                    int M, int N, int K, int relu, int transB, bool big) {
        if (big) {
            // big path requires M%128==0, K%32==0; pads N to multiple of 64
            int Npad = (N + 63) & ~63;
            int na = M * K, nb = Npad * K;
            ggnn_f32_to_bf16<<<ew(na), 256, 0, stream>>>(A, Abf, na);
            ggnn_wconv<<<ew(nb), 256, 0, stream>>>(B, Bbf, K, N, Npad, transB);
            dim3 g(Npad / 64, M / 128);
            ggnn_wmma_gemm<<<g, 128, 0, stream>>>(Abf, Bbf, bias, C, M, N, K, relu);
        } else {
            size_t tot = (size_t)M * N;
            ggnn_naive_gemm<<<(unsigned)((tot + 255) / 256), 256, 0, stream>>>(
                A, B, bias, C, M, N, K, relu, transB);
        }
    };

    // ---- precompute (once per launch; deterministic) ----
    ggnn_naive_gemm<<<ew(NOE * H).x, 256, 0, stream>>>(emb_ent, Wie, bie, nodes_oe, NOE, H, EMB, 0, 0);
    ggnn_naive_gemm<<<ew(NOP * H).x, 256, 0, stream>>>(emb_pred, Wip, bip, nodes_op, NOP, H, EMB, 0, 0);
    hipMemcpyAsync(nodes_ie, obj_fm, (size_t)NE * H * 4, hipMemcpyDeviceToDevice, stream);
    hipMemcpyAsync(nodes_ip, vr, (size_t)NP * H * 4, hipMemcpyDeviceToDevice, stream);
    hipMemsetAsync(img2ont, 0, (size_t)NP * NOP * 4, stream);
    hipMemsetAsync(cnt_s, 0, NE * 4, stream);
    hipMemsetAsync(cnt_o, 0, NE * 4, stream);
    hipMemsetAsync(colsum, 0, NOE * 4, stream);
    ggnn_counts<<<ew(NP), 256, 0, stream>>>(rel, cnt_s, cnt_o, NP);
    ggnn_topk<<<ew(NE), 256, 0, stream>>>(logits, idx5, val5, colsum, NE);
    ggnn_colsum_den<<<ew(4 * NOE), 256, 0, stream>>>(adj_e2e, den_e2e, 4, NOE, NOE);
    ggnn_colsum_den<<<ew(2 * NOP), 256, 0, stream>>>(adj_e2p, den_e2p, 2, NOE, NOP);
    ggnn_colsum_den<<<ew(2 * NOE), 256, 0, stream>>>(adj_p2e, den_p2e, 2, NOP, NOE);
    ggnn_colsum_den<<<ew(2 * NOP), 256, 0, stream>>>(adj_p2p, den_p2p, 2, NOP, NOP);

    auto gru = [&](float* h, const float* m, int g, int M,
                   float* A, float* B, float* C, float* D, bool big) {
        const float* W = gruW + (size_t)g * 6 * H * H;
        const float* b = gruB + (size_t)g * 6 * H;
        int n = M * H;
        gemm(m, W + 0 * (size_t)H * H, b + 0 * H, A, M, H, H, 0, 0, big);
        gemm(h, W + 1 * (size_t)H * H, b + 1 * H, B, M, H, H, 0, 0, big);
        ggnn_sigadd<<<ew(n), 256, 0, stream>>>(A, B, A, n);               // z
        gemm(m, W + 2 * (size_t)H * H, b + 2 * H, B, M, H, H, 0, 0, big);
        gemm(h, W + 3 * (size_t)H * H, b + 3 * H, C, M, H, H, 0, 0, big);
        ggnn_sigadd<<<ew(n), 256, 0, stream>>>(B, C, B, n);               // r
        gemm(m, W + 4 * (size_t)H * H, b + 4 * H, C, M, H, H, 0, 0, big); // X5
        ggnn_mul<<<ew(n), 256, 0, stream>>>(B, h, B, n);                  // r*h
        gemm(B, W + 5 * (size_t)H * H, b + 5 * H, D, M, H, H, 0, 0, big); // X6
        ggnn_grufin<<<ew(n), 256, 0, stream>>>(A, C, D, h, n);
    };

    for (int t = 0; t < TSTEPS; ++t) {
        // ---- send MLPs (relu both layers) ----
        gemm(nodes_oe, sW1 + 0 * (size_t)H * H2, sb1 + 0 * H2, XA, NOE, H2, H, 1, 0, false);
        gemm(XA, sW2 + 0 * (size_t)H2 * H4, sb2 + 0 * H4, ms_oe, NOE, H4, H2, 1, 0, false);
        gemm(nodes_op, sW1 + 1 * (size_t)H * H2, sb1 + 1 * H2, XA, NOP, H2, H, 1, 0, false);
        gemm(XA, sW2 + 1 * (size_t)H2 * H4, sb2 + 1 * H4, ms_op, NOP, H4, H2, 1, 0, false);
        gemm(nodes_ie, sW1 + 2 * (size_t)H * H2, sb1 + 2 * H2, XA, NE, H2, H, 1, 0, true);
        gemm(XA, sW2 + 2 * (size_t)H2 * H4, sb2 + 2 * H4, ms_ie, NE, H4, H2, 1, 0, true);
        gemm(nodes_ip, sW1 + 3 * (size_t)H * H2, sb1 + 3 * H2, S0, NP, H2, H, 1, 0, true);
        gemm(S0, sW2 + 3 * (size_t)H2 * H4, sb2 + 3 * H4, ms_ip, NP, H4, H2, 1, 0, true);

        // ---- incoming messages ----
        hipMemsetAsync(acc_s, 0, (size_t)NE * H4 * 4, stream);
        hipMemsetAsync(acc_o, 0, (size_t)NE * H4 * 4, stream);
        hipMemsetAsync(acc_oe, 0, (size_t)NOE * H4 * 4, stream);
        ggnn_adj_einsum<<<ew(NOE * 4 * 128), 256, 0, stream>>>(adj_e2e, den_e2e, ms_oe, inc_oe, NOE, NOE, 4, 0, 7);
        ggnn_adj_einsum<<<ew(NOE * 2 * 128), 256, 0, stream>>>(adj_p2e, den_p2e, ms_op, inc_oe, NOP, NOE, 2, 4, 7);
        ggnn_adj_einsum<<<ew(NOP * 2 * 128), 256, 0, stream>>>(adj_e2p, den_e2p, ms_oe, inc_op, NOE, NOP, 2, 0, 5);
        ggnn_adj_einsum<<<ew(NOP * 2 * 128), 256, 0, stream>>>(adj_p2p, den_p2p, ms_op, inc_op, NOP, NOP, 2, 2, 5);
        ggnn_scatter_i2o<<<ew(NE * 128), 256, 0, stream>>>(idx5, val5, ms_ie, acc_oe, NE);
        ggnn_scale_acc_oe<<<ew(NOE * 128), 256, 0, stream>>>(acc_oe, colsum, inc_oe);
        ggnn_op_slot4<<<NOP, 128, 0, stream>>>(img2ont, ms_ip, inc_op, NP);
        ggnn_scatter_subobj<<<ew(NP * 128), 256, 0, stream>>>(rel, ms_ip, acc_s, acc_o, NP);
        ggnn_scale_inc_ie<<<ew(NE * 128), 256, 0, stream>>>(acc_s, acc_o, cnt_s, cnt_o, inc_ie, NE);
        ggnn_gather_ie2<<<ew(NE * 128), 256, 0, stream>>>(idx5, val5, ms_oe, inc_ie, NE);
        ggnn_gather_ip01<<<ew(NP * 128), 256, 0, stream>>>(rel, ms_ie, inc_ip, NP);
        ggnn_ip_slot2<<<ew(NP * 128), 256, 0, stream>>>(img2ont, ms_op, inc_ip, NP);
        ggnn_l2norm128<<<NOE * 7, 32, 0, stream>>>(inc_oe, NOE * 7);
        ggnn_l2norm128<<<NOP * 5, 32, 0, stream>>>(inc_op, NOP * 5);
        ggnn_l2norm128<<<NE * 3, 32, 0, stream>>>(inc_ie, NE * 3);
        ggnn_l2norm128<<<NP * 3, 32, 0, stream>>>(inc_ip, NP * 3);

        // ---- receive MLPs (relu both layers) ----
        gemm(inc_oe, roeW1, roeB1, XA, NOE, 7 * H4, 7 * H4, 1, 0, false);
        gemm(XA, roeW2, roeB2, mr_oe, NOE, H, 7 * H4, 1, 0, false);
        gemm(inc_op, ropW1, ropB1, XA, NOP, 5 * H4, 5 * H4, 1, 0, false);
        gemm(XA, ropW2, ropB2, mr_op, NOP, H, 5 * H4, 1, 0, false);
        gemm(inc_ie, rieW1, rieB1, XA, NE, 3 * H4, 3 * H4, 1, 0, true);
        gemm(XA, rieW2, rieB2, mr_ie, NE, H, 3 * H4, 1, 0, true);
        gemm(inc_ip, ripW1, ripB1, S1, NP, 3 * H4, 3 * H4, 1, 0, true);
        gemm(S1, ripW2, ripB2, S0, NP, H, 3 * H4, 1, 0, true);     // mr_ip -> S0

        // ---- GRU updates ----
        gru(nodes_oe, mr_oe, 0, NOE, XA, XB, XC, XD, false);
        gru(nodes_op, mr_op, 1, NOP, XA, XB, XC, XD, false);
        gru(nodes_ie, mr_ie, 2, NE, XA, XB, XC, XD, true);
        gru(nodes_ip, S0, 3, NP, S1, S2, S3, S0, true);            // X6 reuses S0 after m dead

        // ---- output projections + logits ----
        gemm(nodes_op, ooW1, ooB1, XA, NOP, H, H, 1, 0, false);
        gemm(XA, ooW2, ooB2, XB, NOP, H, H, 0, 0, false);          // proj_op [51,512]
        gemm(nodes_ip, oiW1, oiB1, S1, NP, H, H, 1, 0, true);
        gemm(S1, oiW2, oiB2, S2, NP, H, H, 0, 0, true);            // proj_ip [NP,512]
        gemm(S2, XB, nullptr, out, NP, NOP, H, 0, 1, true);        // logits = proj_ip @ proj_op^T
        ggnn_rowsoftmax<<<ew(NP), 256, 0, stream>>>(out, img2ont, NP, NOP);
    }
}
